// DeepVCP_45999099740711
// MI455X (gfx1250) — matrix operand order repaired and checked by reading, not compiled, not run on hardware
//
#include <hip/hip_runtime.h>

// ---------------- problem constants (from reference) ----------------
#define B_    2
#define N_    1024
#define K_    64
#define NS_   32
#define NV_   11
#define C_    1331        // NV^3
#define CPAD_ 1344        // 84 * 16 (candidate tiles of 16)
#define F_    32
#define RADIUS2 1.0f
#define RVOX  2.0f
#define SVOX  0.4f

typedef __attribute__((ext_vector_type(16))) _Float16 v16h;
typedef __attribute__((ext_vector_type(8)))  float    v8f;

// ------------- CDNA5 wave32 WMMA helpers ----------------------------
// A (16x32 f16, MxK) per-lane layout (ISA 7.12.2): lane L holds row
// m=L&15; halves 2v,2v+1 hold K = ((v<4)?2v:2v+8) + 8*(L>=16).
// Inverse map: element (m,k) -> lane = (m&15) + 16*((k>>3)&1),
//                              half = (k&7) + ((k>>1)&8).
__device__ __forceinline__ int fa_lane(int m, int k) {
  return (m & 15) + (((k >> 3) & 1) << 4);
}
__device__ __forceinline__ int fa_half(int k) {
  return (k & 7) + ((k >> 1) & 8);
}
// A fragment = one contiguous 32B chunk per lane (2x ds_load_b128).
__device__ __forceinline__ v16h fragA_lds(const _Float16* buf, int lane) {
  return *(const v16h*)(buf + lane * 16);
}
// B (32x16 f16, KxN): lane L holds column n=L&15; half h holds
// K = ((h<8)?h:h+8) + 8*(L>=16).  Built once per kernel straight from global f32
// (rows >= `rows` are zero padding).
__device__ __forceinline__ v16h fragB_g(const float* Bm, int ldb, int rows,
                                        int lane) {
  int n = lane & 15, hi = lane >> 4;
  v16h b;
#pragma unroll
  for (int h = 0; h < 16; ++h) {
    int r = ((h < 8) ? h : h + 8) + hi * 8;
    b[h] = (_Float16)((r < rows) ? Bm[r * ldb + n] : 0.f);
  }
  return b;
}
__device__ __forceinline__ v8f wmma_f16(v16h a, v16h b, v8f c) {
  return __builtin_amdgcn_wmma_f32_16x16x32_f16(false, a, false, b, (short)0, c,
                                                false, false);
}

// =====================================================================
// Kernel 1: feature extraction (6->32->32 MLP) for src/tgt + score head
// grid (N/128, B, 2[src|tgt]), block 256 (8 waves x 16 points each).
// Layer-1 A fragment is built directly from global (only 6 of 32 K
// columns are non-zero); weights live in VGPR fragments.
// =====================================================================
__global__ void __launch_bounds__(256) k_feat(
    const float* __restrict__ pts_src, const float* __restrict__ pts_tgt,
    const float* __restrict__ W1, const float* __restrict__ b1,
    const float* __restrict__ W2, const float* __restrict__ b2,
    const float* __restrict__ Ww1, const float* __restrict__ bw1,
    const float* __restrict__ Ww2, const float* __restrict__ bw2,
    _Float16* __restrict__ feat_src, _Float16* __restrict__ feat_tgt,
    float* __restrict__ score, float* __restrict__ tgt_xyz) {
  __shared__ __align__(32) _Float16 Hfrag[8][512];  // per-wave A stage
  __shared__ float Sst[8][16 * 16];                 // per-wave score stage

  const int tid = threadIdx.x, lane = tid & 31, wid = tid >> 5;
  const int b = blockIdx.y, side = blockIdx.z;
  const float* pts = side ? pts_tgt : pts_src;
  _Float16* feat   = side ? feat_tgt : feat_src;

  const int p0 = blockIdx.x * 128 + wid * 16;
  const int m = lane & 15, n0 = lane & 15, hi = lane >> 4;

  // weight fragments (VGPR-resident, loaded once from global f32)
  v16h bW1n0 = fragB_g(W1, F_, 6, lane);
  v16h bW1n1 = fragB_g(W1 + 16, F_, 6, lane);
  v16h bW2n0 = fragB_g(W2, F_, 32, lane);
  v16h bW2n1 = fragB_g(W2 + 16, F_, 32, lane);
  const float b1a = b1[n0], b1b = b1[n0 + 16];
  const float b2a = b2[n0], b2b = b2[n0 + 16];

  // layer-1 A fragment straight from global (cols >= 6 are zero)
  v16h a;
#pragma unroll
  for (int v = 0; v < 8; ++v) {
    int kk = ((v < 4) ? (2 * v) : (2 * v + 8)) + hi * 8;
    float x0 = (kk < 6)     ? pts[(size_t)b * 6 * N_ + (size_t)kk * N_ + (p0 + m)] : 0.f;
    float x1 = (kk + 1 < 6) ? pts[(size_t)b * 6 * N_ + (size_t)(kk + 1) * N_ + (p0 + m)] : 0.f;
    a[2 * v]     = (_Float16)x0;
    a[2 * v + 1] = (_Float16)x1;
  }
  v8f h0 = {}, h1 = {};
  h0 = wmma_f16(a, bW1n0, h0);
  h1 = wmma_f16(a, bW1n1, h1);
#pragma unroll
  for (int r = 0; r < 8; ++r) {
    int mm = r + hi * 8;
    Hfrag[wid][fa_lane(mm, n0) * 16 + fa_half(n0)] =
        (_Float16)fmaxf(h0[r] + b1a, 0.f);
    Hfrag[wid][fa_lane(mm, n0 + 16) * 16 + fa_half(n0 + 16)] =
        (_Float16)fmaxf(h1[r] + b1b, 0.f);
  }
  v16h a2 = fragA_lds(Hfrag[wid], lane);
  v8f f0 = {}, f1 = {};
  f0 = wmma_f16(a2, bW2n0, f0);
  f1 = wmma_f16(a2, bW2n1, f1);
#pragma unroll
  for (int r = 0; r < 8; ++r) {
    int mm = r + hi * 8;
    _Float16 u0 = (_Float16)fmaxf(f0[r] + b2a, 0.f);
    _Float16 u1 = (_Float16)fmaxf(f1[r] + b2b, 0.f);
    feat[((size_t)b * N_ + (p0 + mm)) * F_ + n0]      = u0;
    feat[((size_t)b * N_ + (p0 + mm)) * F_ + n0 + 16] = u1;
    Hfrag[wid][fa_lane(mm, n0) * 16 + fa_half(n0)]           = u0;  // reuse
    Hfrag[wid][fa_lane(mm, n0 + 16) * 16 + fa_half(n0 + 16)] = u1;
  }
  if (side == 0) {  // score head (src only)
    v16h a3 = fragA_lds(Hfrag[wid], lane);
    v16h bw = fragB_g(Ww1, 16, 32, lane);
    v8f s = {};
    s = wmma_f16(a3, bw, s);
    const float bwa = bw1[n0];
#pragma unroll
    for (int r = 0; r < 8; ++r)
      Sst[wid][(r + hi * 8) * 16 + n0] = fmaxf(s[r] + bwa, 0.f);
    if (lane < 16) {
      float sc = bw2[0];
#pragma unroll
      for (int n = 0; n < 16; ++n) sc += Sst[wid][lane * 16 + n] * Ww2[n];
      score[b * N_ + p0 + lane] = sc;
    }
  } else {  // tgt: also emit xyz (B,N,3)
    for (int t = lane; t < 16 * 3; t += 32) {
      int r = t / 3, d = t % 3;
      tgt_xyz[((size_t)b * N_ + (p0 + r)) * 3 + d] =
          pts[(size_t)b * 6 * N_ + (size_t)d * N_ + (p0 + r)];
    }
  }
}

// =====================================================================
// Kernel 2: top-K=64 selection + keypoint xyz.  grid(B), block 256.
// =====================================================================
__global__ void __launch_bounds__(256) k_topk(
    const float* __restrict__ score, const float* __restrict__ src_pts,
    int* __restrict__ key_idx, float* __restrict__ kxyz,
    float* __restrict__ out_kxyz) {
  __shared__ float sc[N_];
  __shared__ float rv[256];
  __shared__ int   ri[256];
  const int tid = threadIdx.x, b = blockIdx.x;
  for (int i = tid; i < N_; i += 256) sc[i] = score[b * N_ + i];
  __syncthreads();
  for (int it = 0; it < K_; ++it) {
    float bv = -3.0e38f; int bi = N_;
    for (int i = tid; i < N_; i += 256) {
      float v = sc[i];
      if (v > bv || (v == bv && i < bi)) { bv = v; bi = i; }
    }
    rv[tid] = bv; ri[tid] = bi;
    __syncthreads();
    for (int s = 128; s > 0; s >>= 1) {
      if (tid < s) {
        float v2 = rv[tid + s]; int i2 = ri[tid + s];
        if (v2 > rv[tid] || (v2 == rv[tid] && i2 < ri[tid])) {
          rv[tid] = v2; ri[tid] = i2;
        }
      }
      __syncthreads();
    }
    if (tid == 0) {
      int sel = ri[0];
      key_idx[b * K_ + it] = sel;
      sc[sel] = -3.0e38f;
      for (int d = 0; d < 3; ++d) {
        float v = src_pts[(size_t)b * 6 * N_ + (size_t)d * N_ + sel];
        kxyz[(b * K_ + it) * 3 + d] = v;
        out_kxyz[(b * K_ + it) * 3 + d] = v;  // output 0: kxyz
      }
    }
    __syncthreads();
  }
}

// =====================================================================
// Kernel 3: radius grouping -> gidx.  grid(B), block K_=64.
// =====================================================================
__global__ void k_group(const float* __restrict__ kxyz, int* __restrict__ gidx) {
  __shared__ float kx[K_ * 3];
  const int b = blockIdx.x, k = threadIdx.x;
  for (int i = k; i < K_ * 3; i += K_) kx[i] = kxyz[b * K_ * 3 + i];
  __syncthreads();
  float x0 = kx[k * 3], y0 = kx[k * 3 + 1], z0 = kx[k * 3 + 2];
  int g[NS_]; int cnt = 0;
  for (int j = 0; j < K_; ++j) {
    float dx = kx[j * 3] - x0, dy = kx[j * 3 + 1] - y0, dz = kx[j * 3 + 2] - z0;
    if (dx * dx + dy * dy + dz * dz <= RADIUS2 && cnt < NS_) g[cnt++] = j;
  }
  int first = g[0];  // k itself always qualifies
  for (int s = cnt; s < NS_; ++s) g[s] = first;
  for (int s = 0; s < NS_; ++s) gidx[(b * K_ + k) * NS_ + s] = g[s];
}

// =====================================================================
// Kernel 4: src branch dfe (35->32->32) + max-pool over NS -> src_emb.
// grid(B*K), block 64 (2 waves, 16 rows each).
// =====================================================================
__global__ void __launch_bounds__(64) k_srcemb(
    const float* __restrict__ kxyz, const int* __restrict__ gidx,
    const int* __restrict__ key_idx, const _Float16* __restrict__ feat_src,
    const float* __restrict__ Wd1, const float* __restrict__ bd1,
    const float* __restrict__ Wd2, const float* __restrict__ bd2,
    float* __restrict__ src_emb) {
  __shared__ __align__(32) _Float16 Afrag[2][2][512];  // [Mtile][kchunk]
  __shared__ __align__(32) _Float16 Hfrag[2][512];
  __shared__ float Emb[NS_ * 32];
  __shared__ int sg[NS_], sorig[NS_];

  const int tid = threadIdx.x, lane = tid & 31, wid = tid >> 5;
  const int b = blockIdx.x >> 6, k = blockIdx.x & 63;
  const int n0 = lane & 15, hi = lane >> 4;

  if (tid < NS_) {
    int gg = gidx[(b * K_ + k) * NS_ + tid];
    sg[tid] = gg;
    sorig[tid] = key_idx[b * K_ + gg];
  }
  __syncthreads();

  // stage A (fragment order): NS_ x 64 = [rel(3) | feats(32) | 0-pad]
  for (int i = tid; i < NS_ * 64; i += 64) {
    int r = i >> 6, c = i & 63;
    float v = 0.f;
    if (c < 3)
      v = kxyz[(b * K_ + sg[r]) * 3 + c] - kxyz[(b * K_ + k) * 3 + c];
    else if (c < 35)
      v = (float)feat_src[((size_t)b * N_ + sorig[r]) * F_ + (c - 3)];
    int mt = r >> 4, mm = r & 15, ch = c >> 5, kk = c & 31;
    Afrag[mt][ch][fa_lane(mm, kk) * 16 + fa_half(kk)] = (_Float16)v;
  }

  // weight fragments (VGPR-resident)
  v16h w1c0n0 = fragB_g(Wd1, 32, 32, lane);
  v16h w1c0n1 = fragB_g(Wd1 + 16, 32, 32, lane);
  v16h w1c1n0 = fragB_g(Wd1 + 32 * 32, 32, 3, lane);       // rows 32..34
  v16h w1c1n1 = fragB_g(Wd1 + 32 * 32 + 16, 32, 3, lane);
  v16h w2n0 = fragB_g(Wd2, 32, 32, lane);
  v16h w2n1 = fragB_g(Wd2 + 16, 32, 32, lane);
  const float bd1a = bd1[n0], bd1b = bd1[n0 + 16];
  const float bd2a = bd2[n0], bd2b = bd2[n0 + 16];
  __syncthreads();

  v16h a0 = fragA_lds(Afrag[wid][0], lane);
  v16h a1 = fragA_lds(Afrag[wid][1], lane);
  v8f acc0 = {}, acc1 = {};
  acc0 = wmma_f16(a0, w1c0n0, acc0);
  acc0 = wmma_f16(a1, w1c1n0, acc0);
  acc1 = wmma_f16(a0, w1c0n1, acc1);
  acc1 = wmma_f16(a1, w1c1n1, acc1);
#pragma unroll
  for (int r = 0; r < 8; ++r) {
    int mm = r + hi * 8;
    Hfrag[wid][fa_lane(mm, n0) * 16 + fa_half(n0)] =
        (_Float16)fmaxf(acc0[r] + bd1a, 0.f);
    Hfrag[wid][fa_lane(mm, n0 + 16) * 16 + fa_half(n0 + 16)] =
        (_Float16)fmaxf(acc1[r] + bd1b, 0.f);
  }
  v16h a2 = fragA_lds(Hfrag[wid], lane);
  v8f e0 = {}, e1 = {};
  e0 = wmma_f16(a2, w2n0, e0);
  e1 = wmma_f16(a2, w2n1, e1);
#pragma unroll
  for (int r = 0; r < 8; ++r) {
    int mm = wid * 16 + r + hi * 8;
    Emb[mm * 32 + n0]      = fmaxf(e0[r] + bd2a, 0.f);
    Emb[mm * 32 + n0 + 16] = fmaxf(e1[r] + bd2b, 0.f);
  }
  __syncthreads();
  if (tid < 32) {
    float mx = -3.0e38f;
    for (int s = 0; s < NS_; ++s) mx = fmaxf(mx, Emb[s * 32 + tid]);
    src_emb[(b * K_ + k) * F_ + tid] = mx;
  }
}

// =====================================================================
// Kernel 5 (fused hot path): per (b,k): NN over LDS-resident tgt_xyz
// (+ prefetch of the matching feature row), candidate dfe via WMMA,
// sim via shfl reduction, softmax, weighted centroid.
// grid(B*K), block 256 (8 waves x 16-cand tiles).
// =====================================================================
__global__ void __launch_bounds__(256) k_vcp(
    const float* __restrict__ kxyz, const float* __restrict__ R_init,
    const float* __restrict__ tgt_xyz, const _Float16* __restrict__ feat_tgt,
    const float* __restrict__ Wd1, const float* __restrict__ bd1,
    const float* __restrict__ Wd2, const float* __restrict__ bd2,
    const float* __restrict__ src_emb, float* __restrict__ out_vcp) {
  __shared__ float txyz[N_ * 3];                      // 12 KB
  __shared__ float simbuf[CPAD_];                     // 5.25 KB
  __shared__ int   nnbuf[CPAD_];                      // 5.25 KB
  __shared__ __align__(32) _Float16 Afrag[8][2][512]; // 16 KB
  __shared__ __align__(32) _Float16 Hfrag[8][512];    // 8 KB
  __shared__ float red[256];                          // 1 KB
  __shared__ float semb[32], skt[3];                  // (~48 KB total)

  const int tid = threadIdx.x, lane = tid & 31, wid = tid >> 5;
  const int b = blockIdx.x >> 6, k = blockIdx.x & 63;
  const int n0 = lane & 15, hi = lane >> 4;

  for (int i = tid; i < N_ * 3; i += 256)
    txyz[i] = tgt_xyz[(size_t)b * N_ * 3 + i];
  if (tid < 32) semb[tid] = src_emb[(b * K_ + k) * F_ + tid];
  if (tid < 3) {
    float acc = 0.f;
    for (int j = 0; j < 3; ++j)
      acc += R_init[tid * 3 + j] * kxyz[(b * K_ + k) * 3 + j];
    skt[tid] = acc;
  }
  // weight fragments (VGPR-resident, hoisted out of the tile loop)
  v16h w1c0n0 = fragB_g(Wd1, 32, 32, lane);
  v16h w1c0n1 = fragB_g(Wd1 + 16, 32, 32, lane);
  v16h w1c1n0 = fragB_g(Wd1 + 32 * 32, 32, 3, lane);
  v16h w1c1n1 = fragB_g(Wd1 + 32 * 32 + 16, 32, 3, lane);
  v16h w2n0 = fragB_g(Wd2, 32, 32, lane);
  v16h w2n1 = fragB_g(Wd2 + 16, 32, 32, lane);
  const float bd1a = bd1[n0], bd1b = bd1[n0 + 16];
  const float bd2a = bd2[n0], bd2b = bd2[n0 + 16];
  __syncthreads();
  const float ktx = skt[0], kty = skt[1], ktz = skt[2];

  // ---- phase 1: nearest neighbour for each of C_ candidates ----
  for (int c = tid; c < CPAD_; c += 256) {
    int bn = 0;
    if (c < C_) {
      int ci = c / 121, cj = (c / 11) % 11, cl = c % 11;
      float cx = ktx + (-RVOX + SVOX * ci);
      float cy = kty + (-RVOX + SVOX * cj);
      float cz = ktz + (-RVOX + SVOX * cl);
      float bd = 3.4e38f;
      for (int n = 0; n < N_; ++n) {
        float dx = cx - txyz[n * 3], dy = cy - txyz[n * 3 + 1],
              dz = cz - txyz[n * 3 + 2];
        float d = __builtin_fmaf(dx, dx, __builtin_fmaf(dy, dy, dz * dz));
        if (d < bd) { bd = d; bn = n; }
      }
      // warm L1/L2 for the phase-2 feature gather (global_prefetch_b8)
      __builtin_prefetch(&feat_tgt[((size_t)b * N_ + bn) * F_], 0, 3);
    }
    nnbuf[c] = bn;
  }
  __syncthreads();

  // ---- phase 2: dfe + sim per 16-candidate tile (wave-parallel) ----
  const float s0 = semb[n0], s1 = semb[n0 + 16];
  for (int t = wid; t < CPAD_ / 16; t += 8) {
    for (int i = lane; i < 16 * 64; i += 32) {
      int r = i >> 6, col = i & 63;  // r = m in tile
      int c = t * 16 + r;
      float v = 0.f;
      if (c < C_) {
        int nn = nnbuf[c];
        if (col < 3) {
          int ci = c / 121, cj = (c / 11) % 11, cl = c % 11;
          float cd = (col == 0) ? ktx + (-RVOX + SVOX * ci)
                   : (col == 1) ? kty + (-RVOX + SVOX * cj)
                                : ktz + (-RVOX + SVOX * cl);
          v = cd - txyz[nn * 3 + col];
        } else if (col < 35) {
          v = (float)feat_tgt[((size_t)b * N_ + nn) * F_ + (col - 3)];
        }
      }
      int ch = col >> 5, kk = col & 31;
      Afrag[wid][ch][fa_lane(r, kk) * 16 + fa_half(kk)] = (_Float16)v;
    }
    v16h a0 = fragA_lds(Afrag[wid][0], lane);
    v16h a1 = fragA_lds(Afrag[wid][1], lane);
    v8f acc0 = {}, acc1 = {};
    acc0 = wmma_f16(a0, w1c0n0, acc0);
    acc0 = wmma_f16(a1, w1c1n0, acc0);
    acc1 = wmma_f16(a0, w1c0n1, acc1);
    acc1 = wmma_f16(a1, w1c1n1, acc1);
#pragma unroll
    for (int r = 0; r < 8; ++r) {
      int mm = r + hi * 8;
      Hfrag[wid][fa_lane(mm, n0) * 16 + fa_half(n0)] =
          (_Float16)fmaxf(acc0[r] + bd1a, 0.f);
      Hfrag[wid][fa_lane(mm, n0 + 16) * 16 + fa_half(n0 + 16)] =
          (_Float16)fmaxf(acc1[r] + bd1b, 0.f);
    }
    v16h a2 = fragA_lds(Hfrag[wid], lane);
    v8f e0 = {}, e1 = {};
    e0 = wmma_f16(a2, w2n0, e0);
    e1 = wmma_f16(a2, w2n1, e1);
    // sim[m] = sum_f relu(emb)[m][f] * src_emb[f]; reduce 16 lanes/half
    float part[8];
#pragma unroll
    for (int r = 0; r < 8; ++r) {
      float u0 = fmaxf(e0[r] + bd2a, 0.f);
      float u1 = fmaxf(e1[r] + bd2b, 0.f);
      part[r] = u0 * s0 + u1 * s1;
    }
#pragma unroll
    for (int off = 1; off < 16; off <<= 1) {
#pragma unroll
      for (int r = 0; r < 8; ++r) part[r] += __shfl_xor(part[r], off, 32);
    }
    if (n0 == 0) {  // lanes 0 and 16 write 8 rows each
#pragma unroll
      for (int r = 0; r < 8; ++r) {
        int mm = r + hi * 8;
        int c = t * 16 + mm;
        simbuf[c] = (c < C_) ? part[r] : -3.0e38f;
      }
    }
  }
  __syncthreads();

  // ---- phase 3: softmax over C_ + weighted centroid ----
  float mx = -3.0e38f;
  for (int c = tid; c < C_; c += 256) mx = fmaxf(mx, simbuf[c]);
  red[tid] = mx;
  __syncthreads();
  for (int s = 128; s > 0; s >>= 1) {
    if (tid < s) red[tid] = fmaxf(red[tid], red[tid + s]);
    __syncthreads();
  }
  mx = red[0];
  __syncthreads();
  float se = 0.f, sx = 0.f, sy = 0.f, sz = 0.f;
  for (int c = tid; c < C_; c += 256) {
    float w = __expf(simbuf[c] - mx);
    int ci = c / 121, cj = (c / 11) % 11, cl = c % 11;
    se += w;
    sx += w * (ktx + (-RVOX + SVOX * ci));
    sy += w * (kty + (-RVOX + SVOX * cj));
    sz += w * (ktz + (-RVOX + SVOX * cl));
  }
  float vals[4] = {se, sx, sy, sz};
  for (int q = 0; q < 4; ++q) {
    red[tid] = vals[q];
    __syncthreads();
    for (int s = 128; s > 0; s >>= 1) {
      if (tid < s) red[tid] += red[tid + s];
      __syncthreads();
    }
    vals[q] = red[0];
    __syncthreads();
  }
  if (tid == 0) {
    float inv = 1.f / vals[0];
    out_vcp[(b * K_ + k) * 3 + 0] = vals[1] * inv;
    out_vcp[(b * K_ + k) * 3 + 1] = vals[2] * inv;
    out_vcp[(b * K_ + k) * 3 + 2] = vals[3] * inv;
  }
}

// =====================================================================
extern "C" void kernel_launch(void* const* d_in, const int* in_sizes, int n_in,
                              void* d_out, int out_size, void* d_ws,
                              size_t ws_size, hipStream_t stream) {
  (void)in_sizes; (void)n_in; (void)out_size; (void)ws_size;
  const float* src_pts = (const float*)d_in[0];
  const float* tgt_pts = (const float*)d_in[1];
  const float* R_init  = (const float*)d_in[2];
  /* d_in[3] = t_init : unused by the reference */
  const float* W1  = (const float*)d_in[4];
  const float* b1  = (const float*)d_in[5];
  const float* W2  = (const float*)d_in[6];
  const float* b2  = (const float*)d_in[7];
  const float* Ww1 = (const float*)d_in[8];
  const float* bw1 = (const float*)d_in[9];
  const float* Ww2 = (const float*)d_in[10];
  const float* bw2 = (const float*)d_in[11];
  const float* Wd1 = (const float*)d_in[12];
  const float* bd1 = (const float*)d_in[13];
  const float* Wd2 = (const float*)d_in[14];
  const float* bd2 = (const float*)d_in[15];

  char* ws = (char*)d_ws;
  size_t off = 0;
  auto alloc = [&](size_t bytes) {
    void* p = ws + off;
    off += (bytes + 255) & ~(size_t)255;
    return p;
  };
  float*    tgt_xyz  = (float*)alloc((size_t)B_ * N_ * 3 * sizeof(float));
  _Float16* feat_src = (_Float16*)alloc((size_t)B_ * N_ * F_ * sizeof(_Float16));
  _Float16* feat_tgt = (_Float16*)alloc((size_t)B_ * N_ * F_ * sizeof(_Float16));
  float*    score    = (float*)alloc((size_t)B_ * N_ * sizeof(float));
  int*      key_idx  = (int*)alloc((size_t)B_ * K_ * sizeof(int));
  float*    kxyz     = (float*)alloc((size_t)B_ * K_ * 3 * sizeof(float));
  int*      gidx     = (int*)alloc((size_t)B_ * K_ * NS_ * sizeof(int));
  float*    src_emb  = (float*)alloc((size_t)B_ * K_ * F_ * sizeof(float));

  float* out_kxyz = (float*)d_out;                    // (B,K,3)
  float* out_vcp  = out_kxyz + (size_t)B_ * K_ * 3;   // (B,K,3)

  k_feat<<<dim3(N_ / 128, B_, 2), 256, 0, stream>>>(
      src_pts, tgt_pts, W1, b1, W2, b2, Ww1, bw1, Ww2, bw2, feat_src, feat_tgt,
      score, tgt_xyz);
  k_topk<<<dim3(B_), 256, 0, stream>>>(score, src_pts, key_idx, kxyz, out_kxyz);
  k_group<<<dim3(B_), K_, 0, stream>>>(kxyz, gidx);
  k_srcemb<<<dim3(B_ * K_), 64, 0, stream>>>(kxyz, gidx, key_idx, feat_src,
                                             Wd1, bd1, Wd2, bd2, src_emb);
  k_vcp<<<dim3(B_ * K_), 256, 0, stream>>>(kxyz, R_init, tgt_xyz, feat_tgt,
                                           Wd1, bd1, Wd2, bd2, src_emb, out_vcp);
}
